// MODEL_ATT_COMP_13855564496925
// MI455X (gfx1250) — matrix operand order repaired and checked by reading, not compiled, not run on hardware
//
#include <hip/hip_runtime.h>

typedef __attribute__((ext_vector_type(16))) __bf16 v16bf;
typedef __attribute__((ext_vector_type(8)))  float  v8f;
typedef __attribute__((vector_size(16)))     int    v4i_;

// ---- CDNA5 async global->LDS path (guarded; falls back to direct loads) ----
#if defined(__HIP_DEVICE_COMPILE__) && defined(__has_builtin)
#if __has_builtin(__builtin_amdgcn_global_load_async_to_lds_b128) && \
    __has_builtin(__builtin_amdgcn_s_wait_asynccnt)
#define USE_ASYNC 1
#endif
#endif
#ifndef USE_ASYNC
#define USE_ASYNC 0
#endif

#define AS1 __attribute__((address_space(1)))
#define AS3 __attribute__((address_space(3)))

namespace {
constexpr int Bsz  = 16;
constexpr int Nseq = 1024;
constexpr int DWd  = 200;
constexpr int DKd  = 128;
constexpr int DVd  = 128;
constexpr int NTd  = 50;
constexpr int BN   = Bsz * Nseq;   // 16384
}

// ---------- helpers ----------

__device__ inline unsigned int f2bf(float f) {
  union { float f; unsigned int u; } c; c.f = f;
  unsigned int u = c.u;
  u += 0x7fffu + ((u >> 16) & 1u);   // round-to-nearest-even
  return u >> 16;
}

__device__ inline unsigned int pack2bf(float lo, float hi) {
  return f2bf(lo) | (f2bf(hi) << 16);
}

__device__ inline float rng01(unsigned int a, unsigned int b) {
  unsigned int x = a * 1664525u + b * 1013904223u + 0x9E3779B9u;
  x ^= x >> 16; x *= 0x7feb352du;
  x ^= x >> 15; x *= 0x846ca68bu;
  x ^= x >> 16;
  return ((float)(x >> 8) + 0.5f) * (1.0f / 16777216.0f);
}

template<int BS>
__device__ inline float block_sum(float v, float* red, int tid) {
  red[tid] = v; __syncthreads();
  for (int s = BS >> 1; s > 0; s >>= 1) {
    if (tid < s) red[tid] = red[tid] + red[tid + s];
    __syncthreads();
  }
  float r = red[0]; __syncthreads();
  return r;
}

template<int BS>
__device__ inline float block_max(float v, float* red, int tid) {
  red[tid] = v; __syncthreads();
  for (int s = BS >> 1; s > 0; s >>= 1) {
    if (tid < s) red[tid] = fmaxf(red[tid], red[tid + s]);
    __syncthreads();
  }
  float r = red[0]; __syncthreads();
  return r;
}

// ---------- generic WMMA bf16 GEMM ----------
// C(M,N) = A(M,K) * op(B) + bias,  op(B)=B^T when transB=1 (B is N x K),
// op(B)=B when transB=0 (B is K x N).  f32 in/out, bf16 multiply, f32 acc.
// K must be even.  asyncOK requires lda%4==0 && ldb%4==0 (16B-aligned rows).
__global__ __launch_bounds__(128) void k_gemm(
    const float* __restrict__ A, const float* __restrict__ Bm,
    const float* __restrict__ bias, float* __restrict__ C,
    int M, int N, int K, int lda, int ldb, int ldc, int transB, int asyncOK)
{
  // unpadded: row stride 64B so each lane fragment run is 16B-aligned contiguous
  __shared__ __align__(16) unsigned short As[64][32];
  __shared__ __align__(16) unsigned short Bs[64][32];
  unsigned int* As32 = (unsigned int*)&As[0][0];  // [64][16] dwords
  unsigned int* Bs32 = (unsigned int*)&Bs[0][0];
#if USE_ASYNC
  __shared__ __align__(16) float stA[64 * 32];    // f32 staging (async deposit)
  __shared__ __align__(16) float stB[64 * 32];
#endif

  const int tid  = threadIdx.x;
  const int lane = tid & 31;
  const int wave = tid >> 5;                 // 0..3
  const int m0 = blockIdx.y * 64;
  const int n0 = blockIdx.x * 64;

  v8f zero = {0.f, 0.f, 0.f, 0.f, 0.f, 0.f, 0.f, 0.f};
  v8f acc[4] = {zero, zero, zero, zero};

  const int l15   = lane & 15;
  const int hi    = lane >> 4;               // 0 or 1
  const int akofs = hi << 3;                 // A: elem offset 0 or 8
  const int bkofs = hi << 4;                 // unused; B fragment also reads at akofs
  const int arow  = (wave << 4) + l15;       // row in 64-row tile

  for (int k0 = 0; k0 < K; k0 += 32) {
#if USE_ASYNC
    if (asyncOK) {
      // ---- async stage raw f32 tiles into LDS (4 x 16B per thread/matrix) ----
      #pragma unroll
      for (int s = 0; s < 4; ++s) {
        int idx = (tid + s * 128) * 4;          // linear float idx, multiple of 4
        {
          int r = idx >> 5, c = idx & 31;
          int gm = m0 + r; gm = (gm > M - 1) ? (M - 1) : gm;   // clamp, keep EXEC full
          long long gofs = (long long)gm * lda + (k0 + c);
          __builtin_amdgcn_global_load_async_to_lds_b128(
              (AS1 v4i_*)(void*)(A + gofs), (AS3 v4i_*)(stA + idx), 0, 0);
        }
        if (transB) {
          int r = idx >> 5, c = idx & 31;
          int gn = n0 + r; gn = (gn > N - 1) ? (N - 1) : gn;
          long long gofs = (long long)gn * ldb + (k0 + c);
          __builtin_amdgcn_global_load_async_to_lds_b128(
              (AS1 v4i_*)(void*)(Bm + gofs), (AS3 v4i_*)(stB + idx), 0, 0);
        } else {
          int kk = idx >> 6, c = idx & 63;
          int gk = k0 + kk; gk = (gk > K - 1) ? (K - 1) : gk;
          long long gofs = (long long)gk * ldb + (n0 + c);
          __builtin_amdgcn_global_load_async_to_lds_b128(
              (AS1 v4i_*)(void*)(Bm + gofs), (AS3 v4i_*)(stB + idx), 0, 0);
        }
      }
      __builtin_amdgcn_s_wait_asynccnt(0);
      __syncthreads();
      // ---- convert staged f32 -> packed bf16 tiles (OOB masked to zero) ----
      for (int i = tid; i < 64 * 16; i += 128) {
        int r = i >> 4, kp = i & 15;
        int gm = m0 + r, gk = k0 + kp * 2;
        float f0 = 0.f, f1 = 0.f;
        if (gm < M && gk < K) { f0 = stA[r * 32 + kp * 2]; f1 = stA[r * 32 + kp * 2 + 1]; }
        As32[i] = pack2bf(f0, f1);
      }
      for (int i = tid; i < 64 * 16; i += 128) {
        int r = i >> 4, kp = i & 15;
        int gn = n0 + r, gk = k0 + kp * 2;
        float f0 = 0.f, f1 = 0.f;
        if (gn < N && gk < K) {
          if (transB) { f0 = stB[r * 32 + kp * 2];   f1 = stB[r * 32 + kp * 2 + 1]; }
          else        { f0 = stB[(kp * 2) * 64 + r]; f1 = stB[(kp * 2 + 1) * 64 + r]; }
        }
        Bs32[i] = pack2bf(f0, f1);
      }
    } else
#endif
    {
      (void)asyncOK;
      // ---- direct path: global f32 loads + packed bf16 dword stores ----
      for (int i = tid; i < 64 * 16; i += 128) {
        int r = i >> 4, kp = i & 15;
        int gm = m0 + r, gk = k0 + kp * 2;
        float f0 = 0.f, f1 = 0.f;
        if (gm < M && gk < K) {
          float2 v2 = *(const float2*)(A + (long long)gm * lda + gk);
          f0 = v2.x; f1 = v2.y;
        }
        As32[i] = pack2bf(f0, f1);
      }
      for (int i = tid; i < 64 * 16; i += 128) {
        int r = i >> 4, kp = i & 15;
        int gn = n0 + r, gk = k0 + kp * 2;
        float f0 = 0.f, f1 = 0.f;
        if (gn < N && gk < K) {
          if (transB) {
            float2 v2 = *(const float2*)(Bm + (long long)gn * ldb + gk);
            f0 = v2.x; f1 = v2.y;
          } else {
            f0 = Bm[(long long)gk * ldb + gn];
            f1 = Bm[(long long)(gk + 1) * ldb + gn];
          }
        }
        Bs32[i] = pack2bf(f0, f1);
      }
    }
    __syncthreads();

    // A fragment: two contiguous 16B runs per lane
    union { uint4 q[2]; v16bf v; } af;
    af.q[0] = *(const uint4*)&As[arow][akofs];
    af.q[1] = *(const uint4*)&As[arow][16 + akofs];
    #pragma unroll
    for (int nt = 0; nt < 4; ++nt) {
      int brow = (nt << 4) + l15;
      union { uint4 q[2]; v16bf v; } bfr;
      bfr.q[0] = *(const uint4*)&Bs[brow][akofs];
      bfr.q[1] = *(const uint4*)&Bs[brow][16 + akofs];
      acc[nt] = __builtin_amdgcn_wmma_f32_16x16x32_bf16(
          false, af.v, false, bfr.v, (short)0, acc[nt], false, false);
    }
    __syncthreads();
  }

  const int rbase = m0 + (wave << 4) + (hi << 3);
  #pragma unroll
  for (int nt = 0; nt < 4; ++nt) {
    int cn = n0 + (nt << 4) + l15;
    if (cn < N) {
      float bv = bias ? bias[cn] : 0.0f;
      #pragma unroll
      for (int r = 0; r < 8; ++r) {
        int gm = rbase + r;
        if (gm < M) C[(long long)gm * ldc + cn] = acc[nt][r] + bv;
      }
    }
  }
}

// ---------- embedding gather ----------
__global__ void k_gather(const int* __restrict__ x, const float* __restrict__ emb,
                         float* __restrict__ xe)
{
  long long i = (long long)blockIdx.x * 256 + threadIdx.x;
  const long long total = (long long)BN * DWd;
  if (i < total) {
    long long r = i / DWd;
    int d = (int)(i % DWd);
    xe[i] = emb[(long long)x[r] * DWd + d];
  }
}

// ---------- attention softmax (normal + no-self) + categorical sample ----------
__global__ __launch_bounds__(256) void k_softmax_row(
    const float* __restrict__ score, float* __restrict__ attw,
    float* __restrict__ attwns, int* __restrict__ samples,
    int bi, float inv_scale)
{
  __shared__ float s[Nseq];
  __shared__ float p[Nseq];
  __shared__ float red[256];
  const int row = blockIdx.x;
  const int tid = threadIdx.x;

  for (int j = tid; j < Nseq; j += 256) s[j] = score[(long long)row * Nseq + j] * inv_scale;
  __syncthreads();

  float m = -3.0e38f;
  for (int j = tid; j < Nseq; j += 256) m = fmaxf(m, s[j]);
  float mx = block_max<256>(m, red, tid);

  float sum = 0.f;
  for (int j = tid; j < Nseq; j += 256) sum += __expf(s[j] - mx);
  float denom = block_sum<256>(sum, red, tid);

  for (int j = tid; j < Nseq; j += 256) {
    float e = __expf(s[j] - mx) / denom;
    p[j] = e;
    attw[(long long)row * Nseq + j] = e;
  }

  float mns = -3.0e38f;
  for (int j = tid; j < Nseq; j += 256) if (j != row) mns = fmaxf(mns, s[j]);
  float mxns = block_max<256>(mns, red, tid);
  float sns = 0.f;
  for (int j = tid; j < Nseq; j += 256) if (j != row) sns += __expf(s[j] - mxns);
  float dns = block_sum<256>(sns, red, tid);
  for (int j = tid; j < Nseq; j += 256)
    attwns[(long long)row * Nseq + j] = (j == row) ? 0.0f : __expf(s[j] - mxns) / dns;

  __syncthreads();
  if (tid == 0) {
    float u = rng01((unsigned)row, (unsigned)bi * 2654435761u);
    float c = 0.f; int sel = Nseq - 1;
    for (int j = 0; j < Nseq; ++j) { c += p[j]; if (u <= c) { sel = j; break; } }
    samples[row] = sel;
  }
}

// ---------- softmax over NT per (b,n) row, in place ----------
__global__ __launch_bounds__(64) void k_softmax50(float* __restrict__ t) {
  __shared__ float red[64];
  const long long row = blockIdx.x;
  const int tid = threadIdx.x;
  float x = (tid < NTd) ? t[row * NTd + tid] : -3.0e38f;
  float mx = block_max<64>(x, red, tid);
  float e = (tid < NTd) ? __expf(x - mx) : 0.f;
  float denom = block_sum<64>(e, red, tid);
  if (tid < NTd) t[row * NTd + tid] = e / denom;
}

// ---------- tl[b,t] = sum_n topic_weight[b,n,t] * T2Lw[n] + T2Lb ----------
__global__ __launch_bounds__(256) void k_tl(const float* __restrict__ tw,
    const float* __restrict__ T2Lw, const float* __restrict__ T2Lb,
    float* __restrict__ topicv)
{
  __shared__ float red[256];
  const int b = blockIdx.x / NTd;
  const int t = blockIdx.x % NTd;
  const int tid = threadIdx.x;
  float sum = 0.f;
  for (int n = tid; n < Nseq; n += 256)
    sum += tw[((long long)(b * Nseq + n)) * NTd + t] * T2Lw[n];
  float s = block_sum<256>(sum, red, tid);
  if (tid == 0) topicv[b * NTd + t] = s + T2Lb[0];
}

// ---------- softmax over NT per batch + recon loss ----------
__global__ __launch_bounds__(64) void k_topic_b(float* __restrict__ topicv,
                                                float* __restrict__ accRecon)
{
  __shared__ float red[64];
  const int b = blockIdx.x;
  const int tid = threadIdx.x;
  float x = (tid < NTd) ? topicv[b * NTd + tid] : -3.0e38f;
  float mx = block_max<64>(x, red, tid);
  float e = (tid < NTd) ? __expf(x - mx) : 0.f;
  float denom = block_sum<64>(e, red, tid);
  float pv = e / denom;
  if (tid < NTd) topicv[b * NTd + tid] = pv;
  float rl = (tid < NTd) ? (-logf(fmaxf(pv, 1e-30f)) * pv) : 0.f;
  float rsum = block_sum<64>(rl, red, tid);
  if (tid == 0) atomicAdd(accRecon, rsum);
}

// ---------- topic_recon[b,n,t] = topic[b,t]*L2Tw[n] + L2Tb[n] ----------
__global__ void k_topic_recon(const float* __restrict__ topicv,
    const float* __restrict__ L2Tw, const float* __restrict__ L2Tb,
    float* __restrict__ trecon)
{
  long long i = (long long)blockIdx.x * 256 + threadIdx.x;
  const long long total = (long long)BN * NTd;
  if (i < total) {
    int t = (int)(i % NTd);
    long long bn = i / NTd;
    int n = (int)(bn % Nseq);
    int b = (int)(bn / Nseq);
    trecon[i] = topicv[b * NTd + t] * L2Tw[n] + L2Tb[n];
  }
}

// ---------- word loss: 1 - cos(word_recon_ns, xe), mean over n, sum over b ----------
__global__ __launch_bounds__(256) void k_word_loss(const float* __restrict__ wrecon,
    const float* __restrict__ xe, float* __restrict__ acc)
{
  __shared__ float red[256];
  const long long r = blockIdx.x;
  const int tid = threadIdx.x;
  float d = 0.f, a2 = 0.f, b2 = 0.f;
  for (int i = tid; i < DWd; i += 256) {
    float a = wrecon[r * DWd + i];
    float b = xe[r * DWd + i];
    d += a * b; a2 += a * a; b2 += b * b;
  }
  float sd = block_sum<256>(d, red, tid);
  float sa = block_sum<256>(a2, red, tid);
  float sb = block_sum<256>(b2, red, tid);
  if (tid == 0) {
    float wp = sd / (fmaxf(sqrtf(sa), 1e-12f) * fmaxf(sqrtf(sb), 1e-12f));
    atomicAdd(acc, (1.0f - wp) * (1.0f / (float)Nseq));
  }
}

// ---------- margin loss ----------
__global__ __launch_bounds__(128) void k_margin(const float* __restrict__ wr,
    const float* __restrict__ vrecon, float* __restrict__ acc)
{
  __shared__ float red[128];
  const int r = blockIdx.x;
  const int b = r >> 10;
  const int n = r & (Nseq - 1);
  const int nsn = (n * 389 + 17) & (Nseq - 1);   // deterministic permutation stand-in
  const int tid = threadIdx.x;
  float dp = 0.f, dn = 0.f, nw = 0.f, nv = 0.f, nn = 0.f;
  for (int i = tid; i < DVd; i += 128) {
    float w  = wr[(long long)r * DVd + i];
    float v  = vrecon[(long long)r * DVd + i];
    float wn = wr[((long long)(b * Nseq + nsn)) * DVd + i];
    dp += w * v; dn += wn * v; nw += w * w; nv += v * v; nn += wn * wn;
  }
  float sdp = block_sum<128>(dp, red, tid);
  float sdn = block_sum<128>(dn, red, tid);
  float snw = block_sum<128>(nw, red, tid);
  float snv = block_sum<128>(nv, red, tid);
  float snn = block_sum<128>(nn, red, tid);
  if (tid == 0) {
    float nvv = fmaxf(sqrtf(snv), 1e-12f);
    float pos = sdp / (fmaxf(sqrtf(snw), 1e-12f) * nvv);
    float neg = sdn / (fmaxf(sqrtf(snn), 1e-12f) * nvv);
    atomicAdd(acc, (1.0f - pos + neg) * (1.0f / (float)Nseq));
  }
}

// ---------- sim loss over sampled topic rows ----------
__global__ __launch_bounds__(64) void k_sim(const float* __restrict__ tw,
    const int* __restrict__ samples, float* __restrict__ acc)
{
  __shared__ float red[64];
  const int r = blockIdx.x;
  const int b = r >> 10;
  const int s = samples[r];
  const int tid = threadIdx.x;
  float d = 0.f, a2 = 0.f, g2 = 0.f;
  if (tid < NTd) {
    float a = tw[(long long)r * NTd + tid];
    float g = tw[((long long)(b * Nseq + s)) * NTd + tid];
    d = a * g; a2 = a * a; g2 = g * g;
  }
  float sd = block_sum<64>(d, red, tid);
  float sa = block_sum<64>(a2, red, tid);
  float sg = block_sum<64>(g2, red, tid);
  if (tid == 0) {
    float sim = sd / (fmaxf(sqrtf(sa), 1e-12f) * fmaxf(sqrtf(sg), 1e-12f));
    atomicAdd(acc, (1.0f - sim) * (1.0f / (float)Nseq));
  }
}

// ---------- Sinkhorn (16x50, 50 iters, lam=10) ----------
__global__ __launch_bounds__(64) void k_sinkhorn(const float* __restrict__ topicv,
                                                 float* __restrict__ sinkOut)
{
  __shared__ float Km[Bsz][NTd];
  __shared__ float Mm[Bsz][NTd];
  __shared__ float uu[Bsz];
  __shared__ float vv[NTd];
  __shared__ float red[64];
  const int tid = threadIdx.x;
  for (int i = tid; i < Bsz * NTd; i += 64) {
    int bi = i / NTd, t = i % NTd;
    float st = topicv[i];
    float m = -logf(st + 1e-6f);
    Mm[bi][t] = m;
    Km[bi][t] = __expf(-10.0f * m);
  }
  if (tid < Bsz) uu[tid] = 1.0f;
  __syncthreads();
  const float a = (float)NTd / (float)Bsz;
  for (int it = 0; it < 50; ++it) {
    if (tid < NTd) {
      float s = 0.f;
      for (int i = 0; i < Bsz; ++i) s += Km[i][tid] * uu[i];
      vv[tid] = 1.0f / s;
    }
    __syncthreads();
    if (tid < Bsz) {
      float s = 0.f;
      for (int j = 0; j < NTd; ++j) s += Km[tid][j] * vv[j];
      uu[tid] = a / s;
    }
    __syncthreads();
  }
  if (tid < NTd) {
    float s = 0.f;
    for (int i = 0; i < Bsz; ++i) s += Km[i][tid] * uu[i];
    vv[tid] = 1.0f / s;
  }
  __syncthreads();
  float accv = 0.f;
  for (int i = tid; i < Bsz * NTd; i += 64) {
    int bi = i / NTd, t = i % NTd;
    accv += uu[bi] * Km[bi][t] * vv[t] * Mm[bi][t];
  }
  float s = block_sum<64>(accv, red, tid);
  if (tid == 0) *sinkOut = s;
}

__global__ void k_init(float* __restrict__ acc) {
  if (threadIdx.x < 8) acc[threadIdx.x] = 0.f;
}

__global__ void k_finalize(const float* __restrict__ acc, float* __restrict__ out) {
  // acc: [0]=word, [1]=margin, [2]=recon, [3]=sim, [4]=sink
  out[0] = acc[0] + acc[1];
  out[1] = acc[1];
  out[2] = acc[0];
  out[3] = acc[2];
  out[4] = acc[3];
  out[5] = acc[4];
}

// ---------- host ----------

extern "C" void kernel_launch(void* const* d_in, const int* in_sizes, int n_in,
                              void* d_out, int out_size, void* d_ws, size_t ws_size,
                              hipStream_t stream) {
  (void)in_sizes; (void)n_in; (void)out_size; (void)ws_size;

  const int*   x    = (const int*)  d_in[0];
  const float* emb  = (const float*)d_in[1];
  const float* Kw   = (const float*)d_in[2];
  const float* Kb   = (const float*)d_in[3];
  const float* Qw   = (const float*)d_in[4];
  const float* Qb   = (const float*)d_in[5];
  const float* Vpw  = (const float*)d_in[6];
  const float* Vpb  = (const float*)d_in[7];
  const float* V2Tw = (const float*)d_in[8];
  const float* V2Tb = (const float*)d_in[9];
  const float* T2Lw = (const float*)d_in[10];
  const float* T2Lb = (const float*)d_in[11];
  const float* L2Tw = (const float*)d_in[12];
  const float* L2Tb = (const float*)d_in[13];
  const float* T2Vw = (const float*)d_in[14];
  const float* T2Vb = (const float*)d_in[15];
  const float* V2Ww = (const float*)d_in[16];
  const float* V2Wb = (const float*)d_in[17];

  float* W = (float*)d_ws;
  size_t off = 0;
  auto alloc = [&](size_t n) { float* p = W + off; off += n; return p; };
  float* xe     = alloc((size_t)BN * DWd);
  float* qb     = alloc((size_t)BN * DKd);
  float* kb     = alloc((size_t)BN * DKd);
  float* vb     = alloc((size_t)BN * DVd);
  float* wr     = alloc((size_t)BN * DVd);
  float* wrns   = alloc((size_t)BN * DVd);
  float* score  = alloc((size_t)Nseq * Nseq);
  float* attw   = alloc((size_t)Nseq * Nseq);
  float* attwns = alloc((size_t)Nseq * Nseq);
  float* tw     = alloc((size_t)BN * NTd);
  float* trecon = alloc((size_t)BN * NTd);
  float* vrecon = alloc((size_t)BN * DVd);
  float* wrecon = alloc((size_t)BN * DWd);
  float* topicv = alloc((size_t)Bsz * NTd);
  int*   smp    = (int*)alloc((size_t)BN);
  float* acc    = alloc(8);

  auto gemm = [&](const float* A, const float* Bm, const float* bias, float* C,
                  int M, int Nn, int K, int lda, int ldb, int ldc, int transB) {
    int asyncOK = ((lda & 3) == 0) && ((ldb & 3) == 0);
    dim3 grid((unsigned)((Nn + 63) / 64), (unsigned)((M + 63) / 64), 1);
    k_gemm<<<grid, 128, 0, stream>>>(A, Bm, bias, C, M, Nn, K, lda, ldb, ldc,
                                     transB, asyncOK);
  };

  k_init<<<1, 32, 0, stream>>>(acc);

  {
    long long total = (long long)BN * DWd;
    k_gather<<<(unsigned)((total + 255) / 256), 256, 0, stream>>>(x, emb, xe);
  }

  // q/k/v projections: (BN x 200) @ (128 x 200)^T
  gemm(xe, Qw,  Qb,  qb, BN, DKd, DWd, DWd, DWd, DKd, 1);
  gemm(xe, Kw,  Kb,  kb, BN, DKd, DWd, DWd, DWd, DKd, 1);
  gemm(xe, Vpw, Vpb, vb, BN, DVd, DWd, DWd, DWd, DVd, 1);

  const float inv_scale = 0.0883883476f;   // 1/sqrt(128)
  for (int bi = 0; bi < Bsz; ++bi) {
    const float* qcur = qb + (size_t)bi * Nseq * DKd;
    const float* kcur = kb + (size_t)bi * Nseq * DKd;
    const float* vcur = vb + (size_t)bi * Nseq * DVd;
    // score = q_b @ k_b^T   (1024x1024, K=128)
    gemm(qcur, kcur, nullptr, score, Nseq, Nseq, DKd, DKd, DKd, Nseq, 1);
    k_softmax_row<<<Nseq, 256, 0, stream>>>(score, attw, attwns,
                                            smp + (size_t)bi * Nseq, bi, inv_scale);
    // word_repre = att_w @ v_b    (1024x128, K=1024)
    gemm(attw,   vcur, nullptr, wr   + (size_t)bi * Nseq * DVd,
         Nseq, DVd, Nseq, Nseq, DVd, DVd, 0);
    gemm(attwns, vcur, nullptr, wrns + (size_t)bi * Nseq * DVd,
         Nseq, DVd, Nseq, Nseq, DVd, DVd, 0);
  }

  // topic pipeline
  gemm(wr, V2Tw, V2Tb, tw, BN, NTd, DVd, DVd, DVd, NTd, 1);
  k_softmax50<<<BN, 64, 0, stream>>>(tw);
  k_tl<<<Bsz * NTd, 256, 0, stream>>>(tw, T2Lw, T2Lb, topicv);
  k_topic_b<<<Bsz, 64, 0, stream>>>(topicv, acc + 2);
  {
    long long total = (long long)BN * NTd;
    k_topic_recon<<<(unsigned)((total + 255) / 256), 256, 0, stream>>>(topicv, L2Tw, L2Tb, trecon);
  }
  gemm(trecon, T2Vw, T2Vb, vrecon, BN, DVd, NTd, NTd, NTd, DVd, 1);

  // reconstructions / losses
  gemm(wrns, V2Ww, V2Wb, wrecon, BN, DWd, DVd, DVd, DVd, DWd, 1);
  k_word_loss<<<BN, 256, 0, stream>>>(wrecon, xe, acc + 0);
  k_margin<<<BN, 128, 0, stream>>>(wr, vrecon, acc + 1);
  k_sim<<<BN, 64, 0, stream>>>(tw, smp, acc + 3);
  k_sinkhorn<<<1, 64, 0, stream>>>(topicv, acc + 4);

  k_finalize<<<1, 1, 0, stream>>>(acc, (float*)d_out);
}
